// CausalSelfAttention_74474732912926
// MI455X (gfx1250) — matrix operand order repaired and checked by reading, not compiled, not run on hardware
//
#include <hip/hip_runtime.h>
#include <stdint.h>

// ---------------------------------------------------------------------------
// Causal self-attention for MI455X (gfx1250, wave32, WMMA).
// fp32 -> bf16 convert, QKV GEMM (64x64/wave WMMA bf16), flash-attention
// (32 queries/wave, WMMA bf16 + fp32 online softmax), out-proj GEMM.
// ---------------------------------------------------------------------------

typedef __attribute__((ext_vector_type(16))) __bf16 v16bf;
typedef __attribute__((ext_vector_type(8)))  float  v8f;

constexpr int Bn  = 4;
constexpr int Tn  = 2048;
constexpr int Cn  = 1024;
constexpr int Hn  = 16;
constexpr int Dhn = 64;
constexpr int BT  = Bn * Tn;      // 8192
constexpr int C3  = 3 * Cn;       // 3072

union Frag { uint32_t u[8]; v16bf v; };

__device__ __forceinline__ uint16_t f2bf(float f) {
  uint32_t u = __float_as_uint(f);
  u += 0x7fffu + ((u >> 16) & 1u);          // round-to-nearest-even
  return (uint16_t)(u >> 16);
}

__device__ __forceinline__ v8f wmma_bf16(const Frag& a, const Frag& b, v8f c) {
  return __builtin_amdgcn_wmma_f32_16x16x32_bf16(false, a.v, false, b.v,
                                                 (short)0, c, false, false);
}

// A fragment (16x32 bf16, M = lane&15): two b128 loads from a K-contiguous row.
__device__ __forceinline__ void loadAfrag(Frag& a, const uint32_t* row, int k0, int lg) {
  uint4 t0 = *(const uint4*)(row + ((k0 + lg * 8) >> 1));
  uint4 t1 = *(const uint4*)(row + ((k0 + 16 + lg * 8) >> 1));
  a.u[0] = t0.x; a.u[1] = t0.y; a.u[2] = t0.z; a.u[3] = t0.w;
  a.u[4] = t1.x; a.u[5] = t1.y; a.u[6] = t1.z; a.u[7] = t1.w;
}

// B fragment (32x16 bf16, N = lane&15) read from K-major ("B-transposed") rows.
__device__ __forceinline__ void loadBfrag(Frag& b, const uint32_t* row, int k0, int lg) {
  uint4 t0 = *(const uint4*)(row + ((k0 + lg * 8) >> 1));
  uint4 t1 = *(const uint4*)(row + ((k0 + 16 + lg * 8) >> 1));
  b.u[0] = t0.x; b.u[1] = t0.y; b.u[2] = t0.z; b.u[3] = t0.w;
  b.u[4] = t1.x; b.u[5] = t1.y; b.u[6] = t1.z; b.u[7] = t1.w;
}

// Register-blocked MTx16 x NTx16 tile: A row-major [M][K], Bt K-major [N][K].
template <int MT, int NT>
__device__ __forceinline__ void gemm_block(const uint32_t* __restrict__ A, int lda,
                                           const uint32_t* __restrict__ Bt, int ldb,
                                           int m0, int n0, int K, int lane,
                                           v8f acc[MT][NT]) {
  const int lm = lane & 15, lg = lane >> 4;
  const uint32_t* aRow[MT];
  const uint32_t* bRow[NT];
#pragma unroll
  for (int mi = 0; mi < MT; ++mi)
    aRow[mi] = A + (((size_t)(m0 + mi * 16 + lm) * (size_t)lda) >> 1);
#pragma unroll
  for (int ni = 0; ni < NT; ++ni)
    bRow[ni] = Bt + (((size_t)(n0 + ni * 16 + lm) * (size_t)ldb) >> 1);
  for (int k0 = 0; k0 < K; k0 += 32) {
    Frag a[MT], b[NT];
#pragma unroll
    for (int mi = 0; mi < MT; ++mi) loadAfrag(a[mi], aRow[mi], k0, lg);
#pragma unroll
    for (int ni = 0; ni < NT; ++ni) loadBfrag(b[ni], bRow[ni], k0, lg);
#pragma unroll
    for (int mi = 0; mi < MT; ++mi)
#pragma unroll
      for (int ni = 0; ni < NT; ++ni)
        acc[mi][ni] = wmma_bf16(a[mi], b[ni], acc[mi][ni]);
  }
}

// ------------------------------ converters ---------------------------------

__global__ void __launch_bounds__(256) cvt_bf16_kernel(
    const float* __restrict__ in, uint16_t* __restrict__ out, int n4) {
  int i = (int)(blockIdx.x * blockDim.x + threadIdx.x);
  int stride = (int)(gridDim.x * blockDim.x);
  const float4* in4 = (const float4*)in;
  uint2* out2 = (uint2*)out;
  for (; i < n4; i += stride) {
    float4 f = in4[i];
    uint2 o;
    o.x = (uint32_t)f2bf(f.x) | ((uint32_t)f2bf(f.y) << 16);
    o.y = (uint32_t)f2bf(f.z) | ((uint32_t)f2bf(f.w) << 16);
    out2[i] = o;
  }
}

// in: fp32 [K][N] row-major  ->  out: bf16 [N][K] (K-major for B fragments)
__global__ void __launch_bounds__(256) transpose_bf16_kernel(
    const float* __restrict__ in, uint16_t* __restrict__ out, int K, int N) {
  int i = (int)(blockIdx.x * blockDim.x + threadIdx.x);
  int stride = (int)(gridDim.x * blockDim.x);
  int total = K * N;
  for (; i < total; i += stride) {
    int k = i / N, n = i - k * N;
    out[(size_t)n * K + k] = f2bf(in[i]);
  }
}

// ------------------------------ QKV GEMM -----------------------------------
// 64x64 tile per wave. Epilogue scatters into Q,K: [b][h][t][d] bf16
// (Q pre-scaled by 1/sqrt(Dh)), V: [b][h][d][t] bf16.

__global__ void __launch_bounds__(256) qkv_gemm_kernel(
    const uint16_t* __restrict__ xb, const uint16_t* __restrict__ wT,
    const float* __restrict__ bias,
    uint16_t* __restrict__ Qh, uint16_t* __restrict__ Kh,
    uint16_t* __restrict__ Vt) {
  const int wave = (int)((blockIdx.x * blockDim.x + threadIdx.x) >> 5);
  const int lane = (int)(threadIdx.x & 31);
  constexpr int TN = C3 / 64;               // 48 tiles along N
  const int tm = wave / TN, tn = wave - tm * TN;
  const int lm = lane & 15, lg = lane >> 4;

  v8f acc[4][4];
#pragma unroll
  for (int ni = 0; ni < 4; ++ni) {
    const float bv = bias[tn * 64 + ni * 16 + lm];
#pragma unroll
    for (int mi = 0; mi < 4; ++mi)
#pragma unroll
      for (int i = 0; i < 8; ++i) acc[mi][ni][i] = bv;
  }

  gemm_block<4, 4>((const uint32_t*)xb, Cn, (const uint32_t*)wT, Cn,
                   tm * 64, tn * 64, Cn, lane, acc);

  const int nb  = tn * 64;                  // 64-aligned => single section/head
  const int sec = nb >> 10;                 // 0:Q 1:K 2:V
  const int h   = (nb & (Cn - 1)) >> 6;
#pragma unroll
  for (int mi = 0; mi < 4; ++mi) {
#pragma unroll
    for (int i = 0; i < 8; ++i) {
      const int m = tm * 64 + mi * 16 + i + 8 * lg;
      const int b = m >> 11, t = m & (Tn - 1);
#pragma unroll
      for (int ni = 0; ni < 4; ++ni) {
        const int d = ni * 16 + lm;
        float v = acc[mi][ni][i];
        if (sec == 0) {
          Qh[((size_t)((b * Hn + h) * Tn + t) << 6) + d] = f2bf(v * 0.125f);
        } else if (sec == 1) {
          Kh[((size_t)((b * Hn + h) * Tn + t) << 6) + d] = f2bf(v);
        } else {
          Vt[((size_t)((b * Hn + h) * Dhn + d) << 11) + t] = f2bf(v);
        }
      }
    }
  }
}

// ---------------------------- flash attention ------------------------------
// One wave owns 32 queries of one (b,h): two 16-row tiles sharing all K/V
// fragments. fp32 online softmax; P staged through LDS (C-layout -> A-frag).

__global__ void __launch_bounds__(128) attn_kernel(
    const uint16_t* __restrict__ Qh, const uint16_t* __restrict__ Kh,
    const uint16_t* __restrict__ Vt, uint16_t* __restrict__ Yb) {
  __shared__ uint16_t Pl[4][32 * 32];       // one 32x32 P tile per wave
  const int wid  = (int)(threadIdx.x >> 5);
  const int lane = (int)(threadIdx.x & 31);
  const int w    = (int)blockIdx.x * 4 + wid;
  constexpr int QT = Tn / 32;               // 64 query tiles per (b,h)
  const int bh = w / QT, qt = w - bh * QT;
  const int b  = bh / Hn, h = bh - b * Hn;
  const int lm = lane & 15, lg = lane >> 4;

  const uint32_t* qB = (const uint32_t*)(Qh + (size_t)bh * Tn * Dhn);
  const uint32_t* kB = (const uint32_t*)(Kh + (size_t)bh * Tn * Dhn);
  const uint32_t* vB = (const uint32_t*)(Vt + (size_t)bh * Dhn * Tn);
  const int qBase = qt * 32;

  // Q fragments: 2 row tiles x 2 k-steps (Q is pre-scaled by 1/sqrt(Dh))
  Frag aq[2][2];
#pragma unroll
  for (int mi = 0; mi < 2; ++mi) {
    const uint32_t* qrow = qB + (((size_t)(qBase + mi * 16 + lm) * Dhn) >> 1);
    loadAfrag(aq[mi][0], qrow, 0, lg);
    loadAfrag(aq[mi][1], qrow, 32, lg);
  }

  v8f o[2][4];
  float rm[2][8], rl[2][8];
#pragma unroll
  for (int mi = 0; mi < 2; ++mi)
#pragma unroll
    for (int i = 0; i < 8; ++i) {
      rm[mi][i] = -__builtin_inff();
      rl[mi][i] = 0.0f;
#pragma unroll
      for (int nd = 0; nd < 4; ++nd) o[mi][nd][i] = 0.0f;
    }

  uint16_t* P = Pl[wid];
  const uint32_t* P32 = (const uint32_t*)P;
  const int nkb = qt + 1;                   // causal: key blocks 0..qt

  for (int kb = 0; kb < nkb; ++kb) {
    const int kBase = kb * 32;

    // K fragments for this 32-key block (shared by both query tiles)
    Frag bk[2][2];                          // [key half][k-step]
    {
      const uint32_t* kr0 = kB + (((size_t)(kBase + lm) * Dhn) >> 1);
      const uint32_t* kr1 = kB + (((size_t)(kBase + 16 + lm) * Dhn) >> 1);
      loadBfrag(bk[0][0], kr0, 0, lg); loadBfrag(bk[0][1], kr0, 32, lg);
      loadBfrag(bk[1][0], kr1, 0, lg); loadBfrag(bk[1][1], kr1, 32, lg);
    }

#pragma unroll
    for (int mi = 0; mi < 2; ++mi) {
      v8f s0, s1;
#pragma unroll
      for (int i = 0; i < 8; ++i) { s0[i] = 0.0f; s1[i] = 0.0f; }
      s0 = wmma_bf16(aq[mi][0], bk[0][0], s0);
      s0 = wmma_bf16(aq[mi][1], bk[0][1], s0);
      s1 = wmma_bf16(aq[mi][0], bk[1][0], s1);
      s1 = wmma_bf16(aq[mi][1], bk[1][1], s1);

      // mask + online softmax (row lives in one 16-lane half-group)
#pragma unroll
      for (int i = 0; i < 8; ++i) {
        const int q = qBase + mi * 16 + i + 8 * lg;
        float v0 = s0[i];
        float v1 = s1[i];
        if (kBase + lm > q)      v0 = -__builtin_inff();
        if (kBase + 16 + lm > q) v1 = -__builtin_inff();
        float bm = fmaxf(v0, v1);
#pragma unroll
        for (int off = 1; off < 16; off <<= 1)
          bm = fmaxf(bm, __shfl_xor(bm, off, 32));
        const float nm = fmaxf(rm[mi][i], bm);
        const float sc = __expf(rm[mi][i] - nm);
        const float p0 = __expf(v0 - nm);
        const float p1 = __expf(v1 - nm);
        float ps = p0 + p1;
#pragma unroll
        for (int off = 1; off < 16; off <<= 1)
          ps += __shfl_xor(ps, off, 32);
        rl[mi][i] = rl[mi][i] * sc + ps;
        rm[mi][i] = nm;
#pragma unroll
        for (int nd = 0; nd < 4; ++nd) o[mi][nd][i] *= sc;
        const int row = mi * 16 + i + 8 * lg;
        P[row * 32 + lm]      = f2bf(p0);
        P[row * 32 + 16 + lm] = f2bf(p1);
      }
    }

    asm volatile("s_wait_dscnt 0" ::: "memory");   // P stores -> P loads

    Frag ap[2];
#pragma unroll
    for (int mi = 0; mi < 2; ++mi)
      loadAfrag(ap[mi], P32 + (size_t)(mi * 16 + lm) * 16, 0, lg);

#pragma unroll
    for (int nd = 0; nd < 4; ++nd) {
      Frag bvf;
      const uint32_t* vrow = vB + ((size_t)(nd * 16 + lm) << 10);  // d * Tn/2
      loadBfrag(bvf, vrow, kBase, lg);
#pragma unroll
      for (int mi = 0; mi < 2; ++mi)
        o[mi][nd] = wmma_bf16(ap[mi], bvf, o[mi][nd]);
    }
  }

  // finalize: divide by row sum, write y[b,t, h*64 + d] as bf16
#pragma unroll
  for (int mi = 0; mi < 2; ++mi)
#pragma unroll
    for (int i = 0; i < 8; ++i) {
      const float inv = 1.0f / rl[mi][i];
      const int t = qBase + mi * 16 + i + 8 * lg;
      uint16_t* yrow = Yb + ((size_t)(b * Tn + t) << 10) + (h << 6);
      yrow[lm]      = f2bf(o[mi][0][i] * inv);
      yrow[16 + lm] = f2bf(o[mi][1][i] * inv);
      yrow[32 + lm] = f2bf(o[mi][2][i] * inv);
      yrow[48 + lm] = f2bf(o[mi][3][i] * inv);
    }
}

// ----------------------------- out projection ------------------------------

__global__ void __launch_bounds__(256) out_gemm_kernel(
    const uint16_t* __restrict__ yb, const uint16_t* __restrict__ wT,
    const float* __restrict__ bias, float* __restrict__ out) {
  const int wave = (int)((blockIdx.x * blockDim.x + threadIdx.x) >> 5);
  const int lane = (int)(threadIdx.x & 31);
  constexpr int TN = Cn / 64;               // 16 tiles along N
  const int tm = wave / TN, tn = wave - tm * TN;
  const int lm = lane & 15, lg = lane >> 4;

  v8f acc[4][4];
#pragma unroll
  for (int ni = 0; ni < 4; ++ni) {
    const float bv = bias[tn * 64 + ni * 16 + lm];
#pragma unroll
    for (int mi = 0; mi < 4; ++mi)
#pragma unroll
      for (int i = 0; i < 8; ++i) acc[mi][ni][i] = bv;
  }

  gemm_block<4, 4>((const uint32_t*)yb, Cn, (const uint32_t*)wT, Cn,
                   tm * 64, tn * 64, Cn, lane, acc);

#pragma unroll
  for (int mi = 0; mi < 4; ++mi)
#pragma unroll
    for (int i = 0; i < 8; ++i) {
      const int m = tm * 64 + mi * 16 + i + 8 * lg;
#pragma unroll
      for (int ni = 0; ni < 4; ++ni)
        out[(size_t)m * Cn + tn * 64 + ni * 16 + lm] = acc[mi][ni][i];
    }
}

// --------------------------------- launch ----------------------------------

extern "C" void kernel_launch(void* const* d_in, const int* in_sizes, int n_in,
                              void* d_out, int out_size, void* d_ws, size_t ws_size,
                              hipStream_t stream) {
  (void)in_sizes; (void)n_in; (void)out_size; (void)ws_size;
  const float* x    = (const float*)d_in[0];
  const float* Wqkv = (const float*)d_in[1];
  const float* bqkv = (const float*)d_in[2];
  const float* Wout = (const float*)d_in[3];
  const float* bout = (const float*)d_in[4];
  float* out = (float*)d_out;

  char* ws = (char*)d_ws;
  size_t off = 0;
  auto carve = [&](size_t bytes) -> void* {
    void* p = ws + off;
    off += (bytes + 255) & ~(size_t)255;
    return p;
  };
  uint16_t* xb    = (uint16_t*)carve((size_t)BT * Cn * 2);   // x, bf16
  uint16_t* wqkvT = (uint16_t*)carve((size_t)C3 * Cn * 2);   // W_qkv^T, bf16
  uint16_t* woutT = (uint16_t*)carve((size_t)Cn * Cn * 2);   // W_out^T, bf16
  uint16_t* Qh    = (uint16_t*)carve((size_t)BT * Cn * 2);   // [b,h,t,d], pre-scaled
  uint16_t* Kh    = (uint16_t*)carve((size_t)BT * Cn * 2);   // [b,h,t,d]
  uint16_t* Vt    = (uint16_t*)carve((size_t)BT * Cn * 2);   // [b,h,d,t]
  uint16_t* Yb    = (uint16_t*)carve((size_t)BT * Cn * 2);   // attn out, bf16

  cvt_bf16_kernel<<<8192, 256, 0, stream>>>(x, xb, (BT * Cn) / 4);
  transpose_bf16_kernel<<<12288, 256, 0, stream>>>(Wqkv, wqkvT, Cn, C3);
  transpose_bf16_kernel<<<4096, 256, 0, stream>>>(Wout, woutT, Cn, Cn);

  // (BT/64)*(C3/64) = 128*48 = 6144 waves, 8 waves/block
  qkv_gemm_kernel<<<768, 256, 0, stream>>>(xb, wqkvT, bqkv, Qh, Kh, Vt);

  // B*H*(T/32) = 4096 waves, 4 waves/block
  attn_kernel<<<1024, 128, 0, stream>>>(Qh, Kh, Vt, Yb);

  // (BT/64)*(Cn/64) = 128*16 = 2048 waves, 8 waves/block
  out_gemm_kernel<<<256, 256, 0, stream>>>(Yb, woutT, bout, out);
}
